// RGCNLayer_26628797235284
// MI455X (gfx1250) — matrix operand order repaired and checked by reading, not compiled, not run on hardware
//
#include <hip/hip_runtime.h>
#include <cstddef>

#define NNODES 10000
#define NEDGES 640000
#define DIN    128
#define NODE_TILES (NNODES / 16)   // 625 exact
#define EDGE_TILES (NEDGES / 16)   // 40000 exact
#define RRELU_SLOPE (11.0f / 48.0f)

typedef __attribute__((ext_vector_type(16))) __bf16 v16bf;
typedef __attribute__((ext_vector_type(8)))  __bf16 v8bf;
typedef __attribute__((ext_vector_type(8)))  float  v8f;

union A16 { v16bf v; __bf16 e[16]; };
union B16 { v16bf v; v8bf h[2]; };

// --- A operand: one 16x32 bf16 k-chunk of a row-major f32 matrix tile.
// ISA 7.12.2 (16-bit A 16x32): lane m=lane&15; lanes 0-15 hold K={0..7,16..23},
// lanes 16-31 hold K={8..15,24..31}. kb = kc*32 + (lane>>4)*8.
__device__ __forceinline__ v16bf load_a_row_f32(const float* __restrict__ row, int kb) {
  float t[16];
  *reinterpret_cast<float4*>(t + 0)  = *reinterpret_cast<const float4*>(row + kb);
  *reinterpret_cast<float4*>(t + 4)  = *reinterpret_cast<const float4*>(row + kb + 4);
  *reinterpret_cast<float4*>(t + 8)  = *reinterpret_cast<const float4*>(row + kb + 16);
  *reinterpret_cast<float4*>(t + 12) = *reinterpret_cast<const float4*>(row + kb + 20);
  A16 a;
#pragma unroll
  for (int j = 0; j < 16; ++j) a.e[j] = (__bf16)t[j];
  return a.v;
}

// --- B operand from LDS (W stored row-major [h][d] as bf16).
// ISA 7.12.2 / 7.12.4 B layout: lane holds column n=lane&15,
// K = (lane>>4)*8 + {0..7, 16..23} within the 32-wide k-chunk.
__device__ __forceinline__ v16bf load_b_lds(const __bf16* Wl, int n, int kb2) {
  B16 b;
  b.h[0] = *reinterpret_cast<const v8bf*>(Wl + n * DIN + kb2);
  b.h[1] = *reinterpret_cast<const v8bf*>(Wl + n * DIN + kb2 + 16);
  return b.v;
}

__device__ __forceinline__ v8f wmma_bf16(v16bf a, v16bf b, v8f c) {
  // (neg_a, A, neg_b, B, c_mod, C, reuse_a, reuse_b)
  return __builtin_amdgcn_wmma_f32_16x16x32_bf16(false, a, false, b, (short)0, c, false, false);
}

// Device-scope non-returning float atomic add (segment_sum across WGPs
// requires SCOPE_DEV; default WGP scope would do the RMW in the WGP cache).
__device__ __forceinline__ void atomic_add_f32_dev(float* p, float v) {
  asm volatile("global_atomic_add_f32 %0, %1, off scope:SCOPE_DEV"
               :: "v"(p), "v"(v) : "memory");
}

__global__ __launch_bounds__(256) void zero_kernel(float* __restrict__ out, int n) {
  int i = blockIdx.x * blockDim.x + threadIdx.x;
  if (i < n) out[i] = 0.0f;
}

__global__ __launch_bounds__(256) void relu_kernel(float* __restrict__ out, int n) {
  int i = blockIdx.x * blockDim.x + threadIdx.x;
  if (i < n) {
    float h = out[i];
    out[i] = (h >= 0.0f) ? h : (RRELU_SLOPE * h);
  }
}

// Stage 1: proj_nodes[n][h] = sum_d node_h[n][d] * W[h][d]  (W_src half)
__global__ __launch_bounds__(256) void proj_nodes_kernel(const float* __restrict__ node_h,
                                                         const float* __restrict__ W,
                                                         float* __restrict__ proj) {
  __shared__ __bf16 Wl[DIN * DIN];
  for (int i = threadIdx.x; i < DIN * DIN; i += blockDim.x) {
    int h = i >> 7, d = i & 127;
    Wl[i] = (__bf16)W[h * 2 * DIN + d];  // W_src = W[:, :128]
  }
  __syncthreads();

  const int lane = threadIdx.x & 31;
  const int col  = lane & 15;
  const int lgrp = lane >> 4;
  const int gwave  = blockIdx.x * (blockDim.x >> 5) + (threadIdx.x >> 5);
  const int nwaves = gridDim.x * (blockDim.x >> 5);

  for (int t = gwave; t < NODE_TILES; t += nwaves) {
    const float* row = node_h + (size_t)(t * 16 + col) * DIN;
    v16bf a[4];
#pragma unroll
    for (int kc = 0; kc < 4; ++kc) a[kc] = load_a_row_f32(row, kc * 32 + lgrp * 8);
#pragma unroll
    for (int nb = 0; nb < 8; ++nb) {
      v8f acc = {};
#pragma unroll
      for (int kc = 0; kc < 4; ++kc)
        acc = wmma_bf16(a[kc], load_b_lds(Wl, nb * 16 + col, kc * 32 + lgrp * 8), acc);
      // C/D layout: vgpr v -> M = v + 8*lgrp, N = nb*16 + col
#pragma unroll
      for (int v = 0; v < 8; ++v)
        proj[(size_t)(t * 16 + v + 8 * lgrp) * DIN + nb * 16 + col] = acc[v];
    }
  }
}

// Stage 2: per 16-edge tile: proj_edge via WMMA, add gathered node projection,
// scale by norm, device-scope atomic scatter-add into out[dst].
// USE_PROJ=true : gsrc = precomputed proj_nodes (gather f32).
// USE_PROJ=false: gsrc = node_h; recompute node projection with extra WMMAs.
template <bool USE_PROJ>
__global__ __launch_bounds__(256) void edge_scatter_kernel(const float* __restrict__ gsrc,
                                                           const float* __restrict__ edge_h,
                                                           const float* __restrict__ norm,
                                                           const float* __restrict__ W,
                                                           const int* __restrict__ src,
                                                           const int* __restrict__ dst,
                                                           float* __restrict__ out) {
  __shared__ __bf16 Wl[(USE_PROJ ? 1 : 2) * DIN * DIN];
  for (int i = threadIdx.x; i < DIN * DIN; i += blockDim.x) {
    int h = i >> 7, d = i & 127;
    Wl[i] = (__bf16)W[h * 2 * DIN + DIN + d];               // W_edge = W[:, 128:]
    if (!USE_PROJ) Wl[DIN * DIN + i] = (__bf16)W[h * 2 * DIN + d];  // W_src
  }
  __syncthreads();

  const int lane = threadIdx.x & 31;
  const int col  = lane & 15;
  const int lgrp = lane >> 4;
  const int gwave  = blockIdx.x * (blockDim.x >> 5) + (threadIdx.x >> 5);
  const int nwaves = gridDim.x * (blockDim.x >> 5);

  for (int t = gwave; t < EDGE_TILES; t += nwaves) {
    const int e0 = t * 16;

    // Per-lane metadata for the 8 edge rows this lane owns in the D tile.
    int sE[8]; int dE[8]; float nm[8];
#pragma unroll
    for (int v = 0; v < 8; ++v) {
      int e = e0 + v + 8 * lgrp;
      if (USE_PROJ) sE[v] = src[e];
      dE[v] = dst[e];
      nm[v] = norm[e];
    }

    const float* erow = edge_h + (size_t)(e0 + col) * DIN;
    v16bf aE[4];
#pragma unroll
    for (int kc = 0; kc < 4; ++kc) aE[kc] = load_a_row_f32(erow, kc * 32 + lgrp * 8);

    v16bf aN[4];
    if (!USE_PROJ) {
      const float* nrow = gsrc + (size_t)src[e0 + col] * DIN;
#pragma unroll
      for (int kc = 0; kc < 4; ++kc) aN[kc] = load_a_row_f32(nrow, kc * 32 + lgrp * 8);
    }

#pragma unroll
    for (int nb = 0; nb < 8; ++nb) {
      const int n = nb * 16 + col;
      v8f accE = {};
#pragma unroll
      for (int kc = 0; kc < 4; ++kc)
        accE = wmma_bf16(aE[kc], load_b_lds(Wl, n, kc * 32 + lgrp * 8), accE);

      v8f accN = {};
      if (!USE_PROJ) {
#pragma unroll
        for (int kc = 0; kc < 4; ++kc)
          accN = wmma_bf16(aN[kc], load_b_lds(Wl + DIN * DIN, n, kc * 32 + lgrp * 8), accN);
      }

#pragma unroll
      for (int v = 0; v < 8; ++v) {
        float pv  = USE_PROJ ? gsrc[(size_t)sE[v] * DIN + n] : accN[v];
        float msg = (pv + accE[v]) * nm[v];
        atomic_add_f32_dev(out + (size_t)dE[v] * DIN + n, msg);
      }
    }
  }
}

extern "C" void kernel_launch(void* const* d_in, const int* in_sizes, int n_in,
                              void* d_out, int out_size, void* d_ws, size_t ws_size,
                              hipStream_t stream) {
  (void)in_sizes; (void)n_in;
  const float* node_h = (const float*)d_in[0];
  const float* edge_h = (const float*)d_in[1];
  const float* norm   = (const float*)d_in[2];
  const float* W      = (const float*)d_in[3];
  const int*   src    = (const int*)d_in[4];
  const int*   dst    = (const int*)d_in[5];
  float* out = (float*)d_out;

  const int n = out_size;  // N_NODES * D_HID
  zero_kernel<<<(n + 255) / 256, 256, 0, stream>>>(out, n);

  const size_t proj_bytes = (size_t)NNODES * DIN * sizeof(float);
  if (ws_size >= proj_bytes) {
    float* proj = (float*)d_ws;
    proj_nodes_kernel<<<80, 256, 0, stream>>>(node_h, W, proj);
    edge_scatter_kernel<true><<<2048, 256, 0, stream>>>(proj, edge_h, norm, W, src, dst, out);
  } else {
    edge_scatter_kernel<false><<<2048, 256, 0, stream>>>(node_h, edge_h, norm, W, src, dst, out);
  }

  relu_kernel<<<(n + 255) / 256, 256, 0, stream>>>(out, n);
}
